// SparseEncoderLayer_57200374448504
// MI455X (gfx1250) — matrix operand order repaired and checked by reading, not compiled, not run on hardware
//
#include <hip/hip_runtime.h>

typedef __attribute__((ext_vector_type(16))) __bf16        v16bf;
typedef __attribute__((ext_vector_type(8)))  float         v8f;
typedef __attribute__((ext_vector_type(4)))  unsigned int  v4u;
typedef __attribute__((ext_vector_type(4)))  unsigned int  u32x4;
typedef __attribute__((ext_vector_type(8)))  int           i32x8;
typedef __attribute__((ext_vector_type(4)))  int           i32x4;

union Frag16 { v4u q[2]; v16bf v; };

#if defined(__has_builtin)
#  if __has_builtin(__builtin_amdgcn_tensor_load_to_lds) && \
      __has_builtin(__builtin_amdgcn_s_wait_tensorcnt)
#    define HAVE_TDM 1
#  endif
#endif
#ifndef HAVE_TDM
#  define HAVE_TDM 0
#endif
#if __has_include(<hip/amd_detail/amd_gfx1250_TDM.h>)
#  define TDM_ARITY 6
#else
#  define TDM_ARITY 5
#endif

__device__ __forceinline__ unsigned short f2bf(float x) {
    unsigned u = __float_as_uint(x);
    unsigned r = u + 0x7FFFu + ((u >> 16) & 1u);   // round-to-nearest-even
    return (unsigned short)(r >> 16);
}

#if HAVE_TDM
// Issue a TDM load of a dense 1-D dword array (ndwords) global -> LDS.
// D# per CDNA5 ISA ch.8: 2-D tensor, data_size=4B, tile = whole array.
__device__ __forceinline__ void tdm_load_1d(unsigned lds_off, const void* gptr,
                                            int ndwords) {
    unsigned long long ga = (unsigned long long)(size_t)gptr;
    u32x4 g0 = { 1u,                                   // count=1 (valid D#)
                 lds_off,                              // lds_addr (bytes)
                 (unsigned)ga,                         // global_addr[31:0]
                 ((unsigned)(ga >> 32) & 0x01FFFFFFu)  // global_addr[56:32]
                     | (2u << 30) };                   // type=2 ("image")
    i32x8 g1 = { 0x00020000,                           // data_size=2 (4B)
                 (ndwords & 0xFFFF) << 16,             // tensor_dim0[15:0]
                 ((ndwords >> 16) & 0xFFFF) | (1 << 16), // td0[31:16], td1=1
                 (ndwords & 0xFFFF) << 16,             // tile_dim0
                 1,                                    // tile_dim1=1, tile_dim2=0
                 ndwords,                              // tensor_dim0_stride[31:0]
                 0, 0 };
    i32x4 z4 = { 0, 0, 0, 0 };
#if TDM_ARITY == 6
    i32x8 z8 = { 0, 0, 0, 0, 0, 0, 0, 0 };
    __builtin_amdgcn_tensor_load_to_lds(g0, g1, z4, z4, z8, 0);
#else
    __builtin_amdgcn_tensor_load_to_lds(g0, g1, z4, z4, 0);
#endif
}
#endif

// ---------------------------------------------------------------------------
// Pack fp32 weights [K][32][32] (k, c_in, c_out) into bf16 B-fragment layout:
// per (k, ntile, lane) 8 dwords; lane=(n,hi): dword i = {B[16*hi+2i][16t+n],
// B[16*hi+2i+1][16t+n]} packed lo/hi.  1KB per (k,t).
// ---------------------------------------------------------------------------
__global__ __launch_bounds__(64)
void pack_w_kernel(const float* __restrict__ W, unsigned int* __restrict__ wp, int K) {
    int id = blockIdx.x * 64 + threadIdx.x;
    if (id >= K * 64) return;
    int lane = id & 31;
    int t    = (id >> 5) & 1;
    int k    = id >> 6;
    int n  = lane & 15;
    int hi = lane >> 4;
    unsigned int* o = wp + ((size_t)(k * 2 + t) * 32 + lane) * 8;
#pragma unroll
    for (int i = 0; i < 8; ++i) {
        int kin0 = hi * 16 + 2 * i;
        float x0 = W[((size_t)k * 32 + kin0)     * 32 + t * 16 + n];
        float x1 = W[((size_t)k * 32 + kin0 + 1) * 32 + t * 16 + n];
        o[i] = (unsigned int)f2bf(x0) | ((unsigned int)f2bf(x1) << 16);
    }
}

__global__ __launch_bounds__(256)
void zero_kernel(float* __restrict__ p, int n) {
    int i = blockIdx.x * 256 + threadIdx.x;
    if (i < n) p[i] = 0.0f;
}

// ---------------------------------------------------------------------------
// conv1: feats [N,3] fp32, 27-offset rule-book, W1 [27,3,32] -> x1 [N,32] fp32
// One thread per row; W1 staged in LDS (27*96 floats = 10.1KB).
// ---------------------------------------------------------------------------
__global__ __launch_bounds__(256)
void conv1_kernel(const float* __restrict__ feats, const int* __restrict__ nbr,
                  const float* __restrict__ W1, float* __restrict__ x1, int N) {
    __shared__ float w[27 * 96];
    for (int i = threadIdx.x; i < 27 * 96; i += 256) w[i] = W1[i];
    __syncthreads();
    int row = blockIdx.x * 256 + threadIdx.x;
    if (row >= N) return;
    float acc[32];
#pragma unroll
    for (int d = 0; d < 32; ++d) acc[d] = 0.0f;
    for (int k = 0; k < 27; ++k) {
        int nb = nbr[(size_t)row * 27 + k];
        if (nb < 0) continue;
        float f0 = feats[(size_t)nb * 3 + 0];
        float f1 = feats[(size_t)nb * 3 + 1];
        float f2 = feats[(size_t)nb * 3 + 2];
        const float* wk = &w[k * 96];
#pragma unroll
        for (int d = 0; d < 32; ++d)
            acc[d] += f0 * wk[d] + f1 * wk[32 + d] + f2 * wk[64 + d];
    }
#pragma unroll
    for (int d = 0; d < 32; ++d) x1[(size_t)row * 32 + d] = acc[d];
}

// ---------------------------------------------------------------------------
// WMMA gather-conv. src: bf16 rows [nsrc,32] (64B). idx: rule-book [nrows,K].
// wp: packed B fragments [K][2][32][8] dwords. dst: fp32 [nrows,32].
//
// Block = 256 threads = 8 waves; each wave owns 32 rows (two 16-row M-tiles).
// Packed weights (K*2048B) and the block's 256xK rule-book indices (K*1024B)
// are staged in LDS via the Tensor Data Mover (wave 0 issues two
// TENSOR_LOAD_TO_LDS, waits on TENSORcnt, block barrier publishes). Inner
// loop: 2 gathered A fragments x 2 LDS B fragments -> 4 WMMAs per k, with
// global_prefetch of the k+1 neighbor rows issued under the WMMAs.
// ---------------------------------------------------------------------------
template <int K>
__global__ __launch_bounds__(256)
void gconv_wmma_kernel(const unsigned short* __restrict__ src,
                       const int* __restrict__ idx,
                       const unsigned int* __restrict__ wp,
                       float* __restrict__ dst, int nrows) {
    extern __shared__ unsigned int smem[];
    unsigned int* lw   = smem;                   // K*512 dwords (B fragments)
    int*          sidx = (int*)(smem + K * 512); // 256*K dwords (rule-book)

    const int tid   = threadIdx.x;
    const int wave  = tid >> 5;
    const int lane  = tid & 31;
    const int rbase = blockIdx.x * 256;          // block's first row
    const int wrow  = rbase + wave * 32;         // this wave's first row
    const bool fullblk = (rbase + 256) <= nrows;

#if HAVE_TDM
    if (wave == 0) {
        unsigned lds0 = __builtin_amdgcn_groupstaticsize();   // dyn-LDS base
        tdm_load_1d(lds0, wp, K * 512);
        if (fullblk)
            tdm_load_1d(lds0 + (unsigned)K * 2048u,
                        idx + (size_t)rbase * K, K * 256);
        __builtin_amdgcn_s_wait_tensorcnt(0);
    }
    if (!fullblk) {        // tail block: scalar stage with -1 padding
        for (int t = tid; t < 256 * K; t += 256) {
            int r = rbase + t / K;
            sidx[t] = (r < nrows) ? idx[(size_t)r * K + t % K] : -1;
        }
    }
#else
    {   // vectorized manual staging fallback (b128)
        const v4u* wp4 = (const v4u*)wp;
        v4u* lw4 = (v4u*)lw;
        for (int i = tid; i < K * 128; i += 256) lw4[i] = wp4[i];
        if (fullblk) {
            const v4u* ip4 = (const v4u*)(idx + (size_t)rbase * K);
            v4u* si4 = (v4u*)sidx;
            for (int i = tid; i < 64 * K; i += 256) si4[i] = ip4[i];
        } else {
            for (int t = tid; t < 256 * K; t += 256) {
                int r = rbase + t / K;
                sidx[t] = (r < nrows) ? idx[(size_t)r * K + t % K] : -1;
            }
        }
    }
#endif
    __syncthreads();

    if (wrow >= nrows) return;    // wave-uniform: EXEC stays all-ones for WMMA

    const int m  = lane & 15;
    const int hi = lane >> 4;
    const int* si = sidx + wave * 32 * K;

    const v8f z8 = {0.f,0.f,0.f,0.f,0.f,0.f,0.f,0.f};
    v8f acc00 = z8, acc01 = z8, acc10 = z8, acc11 = z8;
    const v4u zz = {0u, 0u, 0u, 0u};

    for (int k = 0; k < K; ++k) {
        int nb0 = si[m * K + k];
        int nb1 = si[(16 + m) * K + k];
        // prefetch next k's neighbor rows into cache while WMMAs run
        if (k + 1 < K) {
            int p0 = si[m * K + k + 1];
            int p1 = si[(16 + m) * K + k + 1];
            if (p0 >= 0) __builtin_prefetch(src + (size_t)p0 * 32, 0, 3);
            if (p1 >= 0) __builtin_prefetch(src + (size_t)p1 * 32, 0, 3);
        }
        // A fragments: lane (m,hi) takes K-chunk [16hi,16hi+16)
        Frag16 a0, a1;
        if (nb0 >= 0) {
            const v4u* p = (const v4u*)(src + (size_t)nb0 * 32);
            a0.q[0] = p[2 * hi]; a0.q[1] = p[2 * hi + 1];
        } else { a0.q[0] = zz; a0.q[1] = zz; }
        if (nb1 >= 0) {
            const v4u* p = (const v4u*)(src + (size_t)nb1 * 32);
            a1.q[0] = p[2 * hi]; a1.q[1] = p[2 * hi + 1];
        } else { a1.q[0] = zz; a1.q[1] = zz; }
        // B fragments from LDS
        Frag16 b0, b1;
        const v4u* w0 = (const v4u*)(lw + ((k * 2 + 0) * 32 + lane) * 8);
        const v4u* w1 = (const v4u*)(lw + ((k * 2 + 1) * 32 + lane) * 8);
        b0.q[0] = w0[0]; b0.q[1] = w0[1];
        b1.q[0] = w1[0]; b1.q[1] = w1[1];

        acc00 = __builtin_amdgcn_wmma_f32_16x16x32_bf16(false, a0.v, false, b0.v,
                                                        (short)0, acc00, false, false);
        acc01 = __builtin_amdgcn_wmma_f32_16x16x32_bf16(false, a0.v, false, b1.v,
                                                        (short)0, acc01, false, false);
        acc10 = __builtin_amdgcn_wmma_f32_16x16x32_bf16(false, a1.v, false, b0.v,
                                                        (short)0, acc10, false, false);
        acc11 = __builtin_amdgcn_wmma_f32_16x16x32_bf16(false, a1.v, false, b1.v,
                                                        (short)0, acc11, false, false);
    }

    // C/D layout: lane (m,hi), VGPR j -> element (row = base + 8*hi + j, col m)
#pragma unroll
    for (int j = 0; j < 8; ++j) {
        int r0 = wrow + hi * 8 + j;          // M-tile 0
        if (r0 < nrows) {
            dst[(size_t)r0 * 32 + m]      = acc00[j];
            dst[(size_t)r0 * 32 + 16 + m] = acc01[j];
        }
        int r1 = wrow + 16 + hi * 8 + j;     // M-tile 1
        if (r1 < nrows) {
            dst[(size_t)r1 * 32 + m]      = acc10[j];
            dst[(size_t)r1 * 32 + 16 + m] = acc11[j];
        }
    }
}

// ---------------------------------------------------------------------------
// BN stats: per-channel sum/sumsq of (a [+ b2]) over nrows. LDS reduce across
// the 8 wave-groups then fp32 global atomics into stats[0:32]=sum,[32:64]=ssq.
// ---------------------------------------------------------------------------
__global__ __launch_bounds__(256)
void bn_stats_kernel(const float* __restrict__ a, const float* __restrict__ b2,
                     float* __restrict__ stats, int nrows) {
    int c = threadIdx.x & 31;
    int g = threadIdx.x >> 5;
    int gid = blockIdx.x * 8 + g;
    int ngr = gridDim.x * 8;
    float s = 0.0f, ss = 0.0f;
    for (int r = gid; r < nrows; r += ngr) {
        float v = a[(size_t)r * 32 + c];
        if (b2) v += b2[(size_t)r * 32 + c];
        s += v; ss += v * v;
    }
    __shared__ float ls[8][32], lq[8][32];
    ls[g][c] = s; lq[g][c] = ss;
    __syncthreads();
    if (g == 0) {
#pragma unroll
        for (int i = 1; i < 8; ++i) { s += ls[i][c]; ss += lq[i][c]; }
        atomicAdd(&stats[c], s);
        atomicAdd(&stats[32 + c], ss);
    }
}

// ---------------------------------------------------------------------------
// BN apply + ReLU: v = relu((a[+b2]) * scale + shift); optional fp32 out and
// optional packed-bf16 out (feeds next WMMA gconv gather).
// ---------------------------------------------------------------------------
__global__ __launch_bounds__(256)
void bn_apply_kernel(const float* __restrict__ a, const float* __restrict__ b2,
                     const float* __restrict__ gamma, const float* __restrict__ beta,
                     const float* __restrict__ stats, int nrows,
                     float* __restrict__ of32, unsigned short* __restrict__ obf) {
    size_t i = (size_t)blockIdx.x * 256 + threadIdx.x;
    if (i >= (size_t)nrows * 32) return;
    int c = (int)(i & 31);
    float inv = 1.0f / (float)nrows;
    float mu  = stats[c] * inv;
    float var = stats[32 + c] * inv - mu * mu;
    float sc  = gamma[c] * rsqrtf(var + 1e-5f);
    float sh  = beta[c] - mu * sc;
    float v = a[i];
    if (b2) v += b2[i];
    v = fmaxf(v * sc + sh, 0.0f);
    if (of32) of32[i] = v;
    if (obf)  obf[i] = f2bf(v);
}

// ---------------------------------------------------------------------------
extern "C" void kernel_launch(void* const* d_in, const int* in_sizes, int n_in,
                              void* d_out, int out_size, void* d_ws, size_t ws_size,
                              hipStream_t stream) {
    const float* feats = (const float*)d_in[0];
    const int*   nbr   = (const int*)  d_in[1];
    const int*   child = (const int*)  d_in[2];
    const float* W1    = (const float*)d_in[3];
    const float* W2a   = (const float*)d_in[4];
    const float* W2b   = (const float*)d_in[5];
    const float* W3    = (const float*)d_in[6];
    const float* g1 = (const float*)d_in[7];  const float* b1 = (const float*)d_in[8];
    const float* g2 = (const float*)d_in[9];  const float* b2 = (const float*)d_in[10];
    const float* g3 = (const float*)d_in[11]; const float* b3 = (const float*)d_in[12];
    const float* g4 = (const float*)d_in[13]; const float* b4 = (const float*)d_in[14];

    const int N = in_sizes[1] / 27;   // active sites
    const int M = in_sizes[2] / 8;    // downsampled sites

    // ---- workspace carve-up (256B aligned) ----
    char* w = (char*)d_ws;
    size_t off = 0;
    auto carve = [&](size_t bytes) {
        void* p = w + off;
        off += (bytes + 255) & ~(size_t)255;
        return p;
    };
    float*          x1   = (float*)         carve((size_t)N * 32 * 4);
    float*          br   = (float*)         carve((size_t)N * 32 * 4);
    unsigned short* hbf  = (unsigned short*)carve((size_t)N * 32 * 2);
    float*          x3   = (float*)         carve((size_t)M * 32 * 4);
    unsigned int*   wp2a = (unsigned int*)  carve((size_t)27 * 2 * 32 * 8 * 4);
    unsigned int*   wp2b = (unsigned int*)  carve((size_t)27 * 2 * 32 * 8 * 4);
    unsigned int*   wp3  = (unsigned int*)  carve((size_t) 8 * 2 * 32 * 8 * 4);
    float*          st   = (float*)         carve(4 * 64 * 4);  // 4 stat sets

    float* out_f  = (float*)d_out;              // [M,32]
    float* ft2_f  = out_f + (size_t)M * 32;     // [N,32]

    // ---- weight packing + stats zeroing ----
    pack_w_kernel<<<27, 64, 0, stream>>>(W2a, wp2a, 27);
    pack_w_kernel<<<27, 64, 0, stream>>>(W2b, wp2b, 27);
    pack_w_kernel<<< 8, 64, 0, stream>>>(W3,  wp3,   8);
    zero_kernel<<<1, 256, 0, stream>>>(st, 256);

    const int STAT_BLOCKS = 512;
    const size_t lds27 = (size_t)27 * 3072;   // 54KB weights + 27KB indices
    const size_t lds8  = (size_t) 8 * 3072;

    // ---- p1: submanifold 3^3, C 3->32 ----
    conv1_kernel<<<(N + 255) / 256, 256, 0, stream>>>(feats, nbr, W1, x1, N);
    bn_stats_kernel<<<STAT_BLOCKS, 256, 0, stream>>>(x1, nullptr, st + 0, N);
    bn_apply_kernel<<<(int)(((size_t)N * 32 + 255) / 256), 256, 0, stream>>>(
        x1, nullptr, g1, b1, st + 0, N, nullptr, hbf);          // h1 (bf16)

    // ---- p2 residual branch: two 3^3 gconvs 32->32 (WMMA) ----
    gconv_wmma_kernel<27><<<(N + 255) / 256, 256, lds27, stream>>>(hbf, nbr, wp2a, br, N);
    bn_stats_kernel<<<STAT_BLOCKS, 256, 0, stream>>>(br, nullptr, st + 64, N);
    bn_apply_kernel<<<(int)(((size_t)N * 32 + 255) / 256), 256, 0, stream>>>(
        br, nullptr, g2, b2, st + 64, N, nullptr, hbf);         // h2 (bf16)

    gconv_wmma_kernel<27><<<(N + 255) / 256, 256, lds27, stream>>>(hbf, nbr, wp2b, br, N);

    // ---- AddTable + BNReLU -> ft2 (fp32 to d_out, bf16 for conv3) ----
    bn_stats_kernel<<<STAT_BLOCKS, 256, 0, stream>>>(x1, br, st + 128, N);
    bn_apply_kernel<<<(int)(((size_t)N * 32 + 255) / 256), 256, 0, stream>>>(
        x1, br, g3, b3, st + 128, N, ft2_f, hbf);

    // ---- p3: 2^3 stride-2 conv 32->32 (WMMA over child rule-book) ----
    gconv_wmma_kernel<8><<<(M + 255) / 256, 256, lds8, stream>>>(hbf, child, wp3, x3, M);
    bn_stats_kernel<<<STAT_BLOCKS, 256, 0, stream>>>(x3, nullptr, st + 192, M);
    bn_apply_kernel<<<(int)(((size_t)M * 32 + 255) / 256), 256, 0, stream>>>(
        x3, nullptr, g4, b4, st + 192, M, out_f, nullptr);
}
